// GCN_DTAware_82755429859988
// MI455X (gfx1250) — matrix operand order repaired and checked by reading, not compiled, not run on hardware
//
#include <hip/hip_runtime.h>
#include <hip/hip_bf16.h>
#include <stdint.h>
#include <stddef.h>

// ---------------- problem constants (match reference) ----------------
#define B_GRAPHS 8192
#define NG       50
#define DEG      8
#define EPG      (NG*DEG)       // 400 edges per graph (before self loops)
#define EALL     (EPG+NG)       // 450 with self loops
#define F_IN     7
#define C1       16
#define C2       25
#define OUTF     50
#define KFC1     (NG*C2)        // 1250
#define KPAD     1280           // 1250 padded to 40*32 for 16x16x32 wmma
#define KSTEPS1  40
#define KSTEPS2  2              // K=50 padded to 64
#define NTILES   4              // N=50 padded to 64

typedef __attribute__((ext_vector_type(16))) __bf16 v16bf;
typedef __attribute__((ext_vector_type(8)))  __bf16 v8bf;
typedef __attribute__((ext_vector_type(8)))  float  v8f;

__device__ __forceinline__ float selu_f(float x) {
  const float a = 1.6732632423543772f, s = 1.0507009873554805f;
  return s * (x > 0.f ? x : a * (__expf(x) - 1.f));
}
// order-preserving float<->uint key for atomicMax-based segment max
__device__ __forceinline__ unsigned fkey(float f) {
  unsigned u = __float_as_uint(f);
  return (u & 0x80000000u) ? ~u : (u | 0x80000000u);
}
__device__ __forceinline__ float keyf(unsigned k) {
  unsigned u = (k & 0x80000000u) ? (k ^ 0x80000000u) : ~k;
  return __uint_as_float(u);
}
// load 16 bf16 in WMMA A-register order: elems 0..7 at p, elems 8..15 at p+16
__device__ __forceinline__ v16bf load_a32(const __bf16* p) {
  union { struct { v8bf lo, hi; } s; v16bf v; } c;
  c.s.lo = *(const v8bf*)p;
  c.s.hi = *(const v8bf*)(p + 16);
  return c.v;
}

// ---------------- kernel 1: per-graph fused GAT x2 ----------------
// one workgroup (256 thr) per graph; everything in LDS; writes selu(h2+b2)
// as a bf16 row of length 1280 (zero padded) into hsbf.
__global__ __launch_bounds__(256) void gat2_kernel(
    const float* __restrict__ x, const long long* __restrict__ ei,
    const float* __restrict__ W1, const float* __restrict__ as1v,
    const float* __restrict__ ad1v, const float* __restrict__ b1v,
    const float* __restrict__ W2, const float* __restrict__ as2v,
    const float* __restrict__ ad2v, const float* __restrict__ b2v,
    __bf16* __restrict__ hsbf)
{
  const int g = blockIdx.x, tid = threadIdx.x;
  const long long nb = (long long)g * NG;
  const long long eb = (long long)g * EPG;
  const long long EDGES = (long long)B_GRAPHS * EPG;   // dst array offset

  __shared__ float sW1[F_IN*C1], sas1[C1], sad1[C1], sb1[C1];
  __shared__ float sW2[C1*C2], sas2[C2], sad2[C2], sb2[C2];
  __shared__ float sx[NG*F_IN];
  __shared__ float h1[NG*C1];
  __shared__ float hsel[NG*C1];
  __shared__ float h2[NG*C2];
  __shared__ float agg[NG*C2];
  __shared__ float asrc[NG], adst[NG], zsum[NG];
  __shared__ unsigned mkey[NG];
  __shared__ float ew[EALL];
  __shared__ unsigned char lsrc[EALL], ldst[EALL];

  for (int i = tid; i < F_IN*C1; i += 256) sW1[i] = W1[i];
  for (int i = tid; i < C1; i += 256) { sas1[i]=as1v[i]; sad1[i]=ad1v[i]; sb1[i]=b1v[i]; }
  for (int i = tid; i < C1*C2; i += 256) sW2[i] = W2[i];
  for (int i = tid; i < C2; i += 256) { sas2[i]=as2v[i]; sad2[i]=ad2v[i]; sb2[i]=b2v[i]; }
  for (int i = tid; i < NG*F_IN; i += 256) sx[i] = x[nb*F_IN + i];
  for (int j = tid; j < EALL; j += 256) {
    int s, d;
    if (j < EPG) { s = (int)(ei[eb + j] - nb); d = (int)(ei[EDGES + eb + j] - nb); }
    else         { s = d = j - EPG; }                  // self loop
    lsrc[j] = (unsigned char)s; ldst[j] = (unsigned char)d;
  }
  __syncthreads();

  // ---- layer 1 ----
  for (int i = tid; i < NG*C1; i += 256) {
    int n = i / C1, c = i % C1; float acc = 0.f;
    #pragma unroll
    for (int k = 0; k < F_IN; ++k) acc += sx[n*F_IN+k] * sW1[k*C1+c];
    h1[i] = acc;
  }
  __syncthreads();
  for (int n = tid; n < NG; n += 256) {
    float a = 0.f, d = 0.f;
    #pragma unroll
    for (int c = 0; c < C1; ++c) { a += h1[n*C1+c]*sas1[c]; d += h1[n*C1+c]*sad1[c]; }
    asrc[n]=a; adst[n]=d; mkey[n]=0u; zsum[n]=0.f;
  }
  for (int i = tid; i < NG*C1; i += 256) agg[i] = 0.f;
  __syncthreads();
  for (int j = tid; j < EALL; j += 256) {
    float e = asrc[lsrc[j]] + adst[ldst[j]];
    e = e > 0.f ? e : 0.2f*e;                          // leaky_relu
    ew[j] = e;
    atomicMax(&mkey[ldst[j]], fkey(e));
  }
  __syncthreads();
  for (int j = tid; j < EALL; j += 256) {
    float p = __expf(ew[j] - keyf(mkey[ldst[j]]));
    ew[j] = p;
    atomicAdd(&zsum[ldst[j]], p);
  }
  __syncthreads();
  for (int j = tid; j < EALL; j += 256) ew[j] = ew[j] / zsum[ldst[j]];
  __syncthreads();
  for (int t = tid; t < EALL*C1; t += 256) {
    int j = t / C1, c = t % C1;
    atomicAdd(&agg[ldst[j]*C1 + c], ew[j] * h1[lsrc[j]*C1 + c]);
  }
  __syncthreads();
  for (int i = tid; i < NG*C1; i += 256) hsel[i] = selu_f(agg[i] + sb1[i % C1]);
  __syncthreads();

  // ---- layer 2 ----
  for (int i = tid; i < NG*C2; i += 256) {
    int n = i / C2, c = i % C2; float acc = 0.f;
    #pragma unroll
    for (int k = 0; k < C1; ++k) acc += hsel[n*C1+k] * sW2[k*C2+c];
    h2[i] = acc;
  }
  __syncthreads();
  for (int n = tid; n < NG; n += 256) {
    float a = 0.f, d = 0.f;
    #pragma unroll
    for (int c = 0; c < C2; ++c) { a += h2[n*C2+c]*sas2[c]; d += h2[n*C2+c]*sad2[c]; }
    asrc[n]=a; adst[n]=d; mkey[n]=0u; zsum[n]=0.f;
  }
  for (int i = tid; i < NG*C2; i += 256) agg[i] = 0.f;
  __syncthreads();
  for (int j = tid; j < EALL; j += 256) {
    float e = asrc[lsrc[j]] + adst[ldst[j]];
    e = e > 0.f ? e : 0.2f*e;
    ew[j] = e;
    atomicMax(&mkey[ldst[j]], fkey(e));
  }
  __syncthreads();
  for (int j = tid; j < EALL; j += 256) {
    float p = __expf(ew[j] - keyf(mkey[ldst[j]]));
    ew[j] = p;
    atomicAdd(&zsum[ldst[j]], p);
  }
  __syncthreads();
  for (int j = tid; j < EALL; j += 256) ew[j] = ew[j] / zsum[ldst[j]];
  __syncthreads();
  for (int t = tid; t < EALL*C2; t += 256) {
    int j = t / C2, c = t % C2;
    atomicAdd(&agg[ldst[j]*C2 + c], ew[j] * h2[lsrc[j]*C2 + c]);
  }
  __syncthreads();

  // selu(h2 + b2) -> bf16 row [g, 0..1279]
  __bf16* row = hsbf + (size_t)g * KPAD;
  for (int i = tid; i < NG*C2; i += 256)
    row[i] = (__bf16)selu_f(agg[i] + sb2[i % C2]);
  for (int i = NG*C2 + tid; i < KPAD; i += 256) row[i] = (__bf16)0.f;
}

// ---------------- kernel 2: pre-swizzle FC weights into WMMA B layout ----
// B reg layout (32x16 bf16): lane L holds col N = L%16, elems e=0..15 are
// K = ks*32 + (L<16?0:8) + e + (e<8?0:8). Zero-pad N>=50 and K beyond real extent.
__global__ __launch_bounds__(256) void swz_kernel(
    const float* __restrict__ fc1_w, const float* __restrict__ fc2_w,
    __bf16* __restrict__ bsw1, __bf16* __restrict__ bsw2)
{
  const int N1 = NTILES*KSTEPS1*32*16;   // 81920
  const int N2 = NTILES*KSTEPS2*32*16;   // 4096
  int idx = blockIdx.x * 256 + threadIdx.x;
  if (idx < N1) {
    int e = idx & 15, lane = (idx >> 4) & 31;
    int ks = (idx >> 9) % KSTEPS1, nt = idx / (512*KSTEPS1);
    int n = nt*16 + (lane & 15);
    int k = ks*32 + ((lane < 16) ? 0 : 8) + e + ((e < 8) ? 0 : 8);
    float v = (n < OUTF && k < KFC1) ? fc1_w[k*OUTF + n] : 0.f;
    bsw1[idx] = (__bf16)v;
  } else if (idx < N1 + N2) {
    int j = idx - N1;
    int e = j & 15, lane = (j >> 4) & 31;
    int ks = (j >> 9) % KSTEPS2, nt = j / (512*KSTEPS2);
    int n = nt*16 + (lane & 15);
    int k = ks*32 + ((lane < 16) ? 0 : 8) + e + ((e < 8) ? 0 : 8);
    float v = (n < OUTF && k < OUTF) ? fc2_w[k*OUTF + n] : 0.f;
    bsw2[j] = (__bf16)v;
  }
}

// ---------------- kernel 3: fused FC1(selu) + FC2(mask) via WMMA ---------
// one wave per 16-row M tile; 4 waves per block; 128 blocks cover 8192 rows.
__global__ __launch_bounds__(128) void fc_kernel(
    const __bf16* __restrict__ hsbf,
    const __bf16* __restrict__ bsw1, const __bf16* __restrict__ bsw2,
    const float* __restrict__ fc1_b, const float* __restrict__ fc2_b,
    const unsigned char* __restrict__ mask,
    float* __restrict__ out)
{
  const int lane  = threadIdx.x & 31, wid = threadIdx.x >> 5;
  const int mtile = blockIdx.x * 4 + wid;
  const int r0    = mtile * 16;
  const int arow  = lane & 15;
  const int khalf = (lane < 16) ? 0 : 8;

  __shared__ __align__(32) __bf16 tbuf[4][16*64];   // per-wave FC1 result

  v8f acc[NTILES] = {};
  const __bf16* abase = hsbf + (size_t)(r0 + arow) * KPAD + khalf;
  for (int ks = 0; ks < KSTEPS1; ++ks) {
    v16bf a = load_a32(abase + ks*32);
    #pragma unroll
    for (int t = 0; t < NTILES; ++t) {
      v16bf b = *(const v16bf*)(bsw1 + ((size_t)(t*KSTEPS1 + ks)*32 + lane)*16);
      acc[t] = __builtin_amdgcn_wmma_f32_16x16x32_bf16(
          false, a, false, b, (short)0, acc[t], false, false);
    }
  }
  // FC1 epilogue: +bias, selu, stash bf16 row-major 16x64 tile in LDS
  #pragma unroll
  for (int t = 0; t < NTILES; ++t) {
    int col = t*16 + (lane & 15);
    float bn = (col < OUTF) ? fc1_b[col] : 0.f;
    #pragma unroll
    for (int r = 0; r < 8; ++r) {
      int row = (lane < 16) ? r : (r + 8);
      tbuf[wid][row*64 + col] = (__bf16)selu_f(acc[t][r] + bn);
    }
  }
  __syncthreads();

  // FC2: K = 64 (2 ksteps), A from LDS in register layout
  v8f acc2[NTILES] = {};
  #pragma unroll
  for (int ks = 0; ks < KSTEPS2; ++ks) {
    v16bf a = load_a32(&tbuf[wid][arow*64 + ks*32 + khalf]);
    #pragma unroll
    for (int t = 0; t < NTILES; ++t) {
      v16bf b = *(const v16bf*)(bsw2 + ((size_t)(t*KSTEPS2 + ks)*32 + lane)*16);
      acc2[t] = __builtin_amdgcn_wmma_f32_16x16x32_bf16(
          false, a, false, b, (short)0, acc2[t], false, false);
    }
  }
  // FC2 epilogue: +bias, apply mask -> -inf, store fp32 output
  #pragma unroll
  for (int t = 0; t < NTILES; ++t) {
    int col = t*16 + (lane & 15);
    if (col < OUTF) {
      float bn = fc2_b[col];
      #pragma unroll
      for (int r = 0; r < 8; ++r) {
        int row = (lane < 16) ? r : (r + 8);
        size_t o = (size_t)(r0 + row) * OUTF + col;
        float v = acc2[t][r] + bn;
        out[o] = mask[o] ? v : -__builtin_inff();
      }
    }
  }
}

// ---------------- launcher ----------------
extern "C" void kernel_launch(void* const* d_in, const int* in_sizes, int n_in,
                              void* d_out, int out_size, void* d_ws, size_t ws_size,
                              hipStream_t stream) {
  const float*         x    = (const float*)d_in[0];
  const long long*     ei   = (const long long*)d_in[1];     // int64 [2,E]
  const unsigned char* mask = (const unsigned char*)d_in[2]; // bool [B,50]
  const float* W1  = (const float*)d_in[3];
  const float* as1 = (const float*)d_in[4];
  const float* ad1 = (const float*)d_in[5];
  const float* b1  = (const float*)d_in[6];
  const float* W2  = (const float*)d_in[7];
  const float* as2 = (const float*)d_in[8];
  const float* ad2 = (const float*)d_in[9];
  const float* b2  = (const float*)d_in[10];
  const float* fc1w = (const float*)d_in[11];
  const float* fc1b = (const float*)d_in[12];
  const float* fc2w = (const float*)d_in[13];
  const float* fc2b = (const float*)d_in[14];
  float* out = (float*)d_out;

  // workspace: bf16 activations [8192,1280] + swizzled FC weights
  __bf16* hsbf = (__bf16*)d_ws;                            // 20.97 MB
  __bf16* bsw1 = hsbf + (size_t)B_GRAPHS * KPAD;           // 160 KB
  __bf16* bsw2 = bsw1 + (size_t)NTILES*KSTEPS1*32*16;      // 8 KB

  const int swz_total = NTILES*KSTEPS1*32*16 + NTILES*KSTEPS2*32*16; // 86016
  swz_kernel<<<(swz_total + 255) / 256, 256, 0, stream>>>(fc1w, fc2w, bsw1, bsw2);
  gat2_kernel<<<B_GRAPHS, 256, 0, stream>>>(x, ei, W1, as1, ad1, b1,
                                            W2, as2, ad2, b2, hsbf);
  fc_kernel<<<B_GRAPHS / 64, 128, 0, stream>>>(hsbf, bsw1, bsw2,
                                               fc1b, fc2b, mask, out);
}
